// SoftmaxMatMulSparse_43508018709064
// MI455X (gfx1250) — compile-verified, matrix operand order checked by reading, not run on hardware
//
#include <hip/hip_runtime.h>

typedef __attribute__((ext_vector_type(16))) _Float16 v16h;
typedef __attribute__((ext_vector_type(8)))  float    v8f;

#define SEQ    2048
#define DH     64
#define MROWS  32            // output rows per workgroup
#define CHUNK  128           // softmax chunk width (columns)
#define NCHUNK (SEQ / CHUNK) // 16
#define KC     128           // WMMA K-chunk staged per barrier round
#define NKC    (SEQ / KC)    // 16 staging rounds
#define BTS    (KC + 8)      // bhT row stride in halves (136 -> bank stride 4, conflict-free)

// dynamic LDS layout (bytes)
#define OFF_PROBS 0
#define SZ_PROBS  (MROWS * SEQ * 2)        // 131072: probs f16 [32][2048]
#define OFF_CHUNK (OFF_PROBS + SZ_PROBS)
#define SZ_CHUNK  (DH * BTS * 2)           // 17408: bhT f16 [64][136]; aliases f32 chunk [32][128]=16384
#define OFF_MCS   (OFF_CHUNK + SZ_CHUNK)
#define SZ_MCS    (MROWS * NCHUNK * 4)     // 2048
#define OFF_LCS   (OFF_MCS + SZ_MCS)
#define SZ_LCS    (MROWS * NCHUNK * 4)     // 2048
#define SMEM_BYTES (OFF_LCS + SZ_LCS)      // 152576 (<160KB -> 2 WGs/WGP)

__global__ __launch_bounds__(256)
void SoftmaxMatMulSparse_43508018709064_kernel(const float* __restrict__ x1,
                                               const float* __restrict__ x2,
                                               float* __restrict__ out) {
  extern __shared__ char smem[];
  _Float16* probs = (_Float16*)(smem + OFF_PROBS);   // [MROWS][SEQ]
  float*    chk   = (float*)   (smem + OFF_CHUNK);   // [MROWS][CHUNK]  (phase 1)
  _Float16* bhT   = (_Float16*)(smem + OFF_CHUNK);   // [DH][BTS]      (phase 2, aliased)
  float*    mcs   = (float*)   (smem + OFF_MCS);     // [MROWS][NCHUNK] chunk-max -> scale
  float*    lcs   = (float*)   (smem + OFF_LCS);     // [MROWS][NCHUNK] chunk-sum

  const int t    = threadIdx.x;
  const int head = blockIdx.y;
  const int row0 = blockIdx.x * MROWS;

  const size_t x1base = ((size_t)head * SEQ + row0) * SEQ;
  const size_t x2base = (size_t)head * SEQ * DH;

  // ---------------- Phase 1: single-read chunked softmax into LDS (f16) ----
  const int r = t >> 3;   // row 0..31 (8 consecutive lanes per row, same wave)
  const int s = t & 7;
  for (int c = 0; c < NCHUNK; ++c) {
    const int cb = c * CHUNK;
    // cooperative load of [32][128] fp32 chunk, fully coalesced float4
    #pragma unroll
    for (int v = 0; v < 4; ++v) {
      int f  = (v * 256 + t) * 4;
      int lr = f >> 7;                 // / CHUNK
      int lc = f & (CHUNK - 1);
      const float* gp = x1 + x1base + (size_t)lr * SEQ + cb + lc;
      *(float4*)(chk + lr * CHUNK + lc) = *(const float4*)gp;
      if (c + 1 < NCHUNK) __builtin_prefetch(gp + CHUNK, 0, 0); // global_prefetch_b8
    }
    __syncthreads();
    // chunk-row max: local 16 + 8-lane shuffle reduce (no barrier)
    float m = -3.4e38f;
    #pragma unroll
    for (int j = 0; j < 16; ++j) m = fmaxf(m, chk[r * CHUNK + s * 16 + j]);
    #pragma unroll
    for (int off = 1; off < 8; off <<= 1) m = fmaxf(m, __shfl_xor(m, off, 32));
    // exp(x - mc) stored as f16 + chunk-row sum via shuffle
    float l = 0.f;
    #pragma unroll
    for (int j = 0; j < 16; ++j) {
      float p = __expf(chk[r * CHUNK + s * 16 + j] - m);
      probs[r * SEQ + cb + s * 16 + j] = (_Float16)p;
      l += p;
    }
    #pragma unroll
    for (int off = 1; off < 8; off <<= 1) l += __shfl_xor(l, off, 32);
    if (s == 0) {
      mcs[r * NCHUNK + c] = m;
      lcs[r * NCHUNK + c] = l;
    }
    __syncthreads();   // chunk buffer reused next iter; mcs/lcs published
  }
  // per-row global max M, sum L; rewrite mcs as per-chunk scale exp(mc-M)/L
  if (t < MROWS) {
    float M = -3.4e38f;
    for (int c = 0; c < NCHUNK; ++c) M = fmaxf(M, mcs[t * NCHUNK + c]);
    float L = 0.f;
    for (int c = 0; c < NCHUNK; ++c) L += lcs[t * NCHUNK + c] * __expf(mcs[t * NCHUNK + c] - M);
    float inv = 1.f / L;
    for (int c = 0; c < NCHUNK; ++c)
      mcs[t * NCHUNK + c] = __expf(mcs[t * NCHUNK + c] - M) * inv;
  }
  __syncthreads();
  // LDS-only normalize pass
  for (int i = 0; i < (MROWS * SEQ / 2) / 256; ++i) {
    int h0  = (i * 256 + t) * 2;
    int lr  = h0 >> 11;                 // / SEQ
    int col = h0 & (SEQ - 1);
    float sc = mcs[lr * NCHUNK + (col >> 7)];
    _Float16* p = probs + h0;
    p[0] = (_Float16)((float)p[0] * sc);
    p[1] = (_Float16)((float)p[1] * sc);
  }
  __syncthreads();

  // ---------------- Phase 2: WMMA  out[32][64] = probs[32][2048] @ x2[2048][64]
  const int wave = t >> 5;     // 8 wave32s
  const int lane = t & 31;
  const int nl = lane & 15;
  const int hi = lane >> 4;
  const int m0 = (wave >> 2) * 16;   // M tile (0 or 16)
  const int n0 = (wave & 3) * 16;    // N tile (0,16,32,48)

  v8f acc = {};
  for (int kb = 0; kb < SEQ; kb += KC) {
    // stage B K-chunk [KC][64] fp32 -> f16 bhT[n][k] (K-contiguous per column)
    #pragma unroll
    for (int i = 0; i < 8; ++i) {
      int f = (i * 256 + t) * 4;       // 2048 float4 = [128][64] floats
      int k = f >> 6;
      int n = f & 63;
      float4 v4 = *(const float4*)(x2 + x2base + (size_t)(kb + k) * DH + n);
      bhT[(n + 0) * BTS + k] = (_Float16)v4.x;
      bhT[(n + 1) * BTS + k] = (_Float16)v4.y;
      bhT[(n + 2) * BTS + k] = (_Float16)v4.z;
      bhT[(n + 3) * BTS + k] = (_Float16)v4.w;
    }
    __syncthreads();
    // 4 WMMAs per staging round (K = kb+kk .. +32)
    #pragma unroll
    for (int kk = 0; kk < KC; kk += 32) {
      union { v16h v; uint4 q[2]; } ua, ub;
      // A tile: lane M = m0+nl; halves j -> K = kb+kk + (j<8 ? j : j+8) + 8*hi
      const _Float16* pa = probs + (m0 + nl) * SEQ + kb + kk + hi * 8;
      ua.q[0] = *(const uint4*)(pa);
      ua.q[1] = *(const uint4*)(pa + 16);
      // B tile: lane N = n0+nl; same K striping, K-contiguous in bhT
      const _Float16* pb = bhT + (n0 + nl) * BTS + kk + hi * 8;
      ub.q[0] = *(const uint4*)(pb);
      ub.q[1] = *(const uint4*)(pb + 16);
      acc = __builtin_amdgcn_wmma_f32_16x16x32_f16(false, ua.v, false, ub.v,
                                                   (short)0, acc, false, false);
    }
    __syncthreads();   // bhT reused next round
  }

  // ---------------- Phase 3: store D (VGPR i -> row m0+i+8*hi, col n0+nl)
  #pragma unroll
  for (int i = 0; i < 8; ++i) {
    int m = m0 + i + hi * 8;
    out[((size_t)(head * SEQ + row0 + m)) * DH + n0 + nl] = acc[i];
  }
}

extern "C" void kernel_launch(void* const* d_in, const int* in_sizes, int n_in,
                              void* d_out, int out_size, void* d_ws, size_t ws_size,
                              hipStream_t stream) {
  (void)in_sizes; (void)n_in; (void)out_size; (void)d_ws; (void)ws_size;
  const float* x1 = (const float*)d_in[0];
  const float* x2 = (const float*)d_in[1];
  float* out = (float*)d_out;
  dim3 grid(SEQ / MROWS, 16);   // 64 row-blocks x 16 heads
  dim3 block(256);              // 8 wave32s
  SoftmaxMatMulSparse_43508018709064_kernel<<<grid, block, SMEM_BYTES, stream>>>(x1, x2, out);
}
